// GraphIdentifierEnv_12575664243494
// MI455X (gfx1250) — matrix operand order repaired and checked by reading, not compiled, not run on hardware
//
#include <hip/hip_runtime.h>
#include <hip/hip_bf16.h>

typedef __attribute__((ext_vector_type(4)))  float  v4f;
typedef __attribute__((ext_vector_type(8)))  float  v8f;
typedef __attribute__((ext_vector_type(8)))  __bf16 v8bf;
typedef __attribute__((ext_vector_type(16))) __bf16 v16bf;

#define NMAT 2048
#define SSTEP 0.1f
#define LDT 40   // padded LDS row stride (ushorts): 80B rows -> conflict-free b128 frag loads

// ---------------------------------------------------------------------------
// fp32 -> (bf16 hi, bf16 lo) split:  x ~= hi + lo, residual ~2^-17 |x|
// ---------------------------------------------------------------------------
__device__ __forceinline__ void split1(float x, unsigned short& hi, unsigned short& lo) {
    unsigned u   = __float_as_uint(x);
    unsigned hib = u & 0xffff0000u;               // truncate to bf16
    hi = (unsigned short)(hib >> 16);
    float r = x - __uint_as_float(hib);           // exact in fp32
    lo = (unsigned short)(__float_as_uint(r) >> 16);
}

__device__ __forceinline__ void split4_store(unsigned short* __restrict__ dhi,
                                             unsigned short* __restrict__ dlo,
                                             int idx, v4f v) {
    unsigned short h0,h1,h2,h3,l0,l1,l2,l3;
    split1(v.x,h0,l0); split1(v.y,h1,l1); split1(v.z,h2,l2); split1(v.w,h3,l3);
    uint2 hp, lp;
    hp.x = (unsigned)h0 | ((unsigned)h1 << 16);
    hp.y = (unsigned)h2 | ((unsigned)h3 << 16);
    lp.x = (unsigned)l0 | ((unsigned)l1 << 16);
    lp.y = (unsigned)l2 | ((unsigned)l3 << 16);
    *(uint2*)(dhi + idx) = hp;   // 8B aligned: 80*row + 8*(c4/4)
    *(uint2*)(dlo + idx) = lp;
}

// two 16B LDS chunks -> one v16bf WMMA fragment
__device__ __forceinline__ v16bf ld_frag(const unsigned short* p, int off0, int off1) {
    v8bf c0 = *(const v8bf*)(p + off0);
    v8bf c1 = *(const v8bf*)(p + off1);
    return __builtin_shufflevector(c0, c1, 0,1,2,3,4,5,6,7,8,9,10,11,12,13,14,15);
}

// ---------------------------------------------------------------------------
// Kernel 1: X = S * A_hat(Theta) - S * I   (row softmax, zero diag, renorm)
// ---------------------------------------------------------------------------
__global__ __launch_bounds__(256)
void xprep_kernel(const float* __restrict__ Theta, float* __restrict__ X) {
    __shared__ float red_max[8];
    __shared__ float red_sum[8];
    const int row  = blockIdx.x;
    const int tid  = threadIdx.x;
    const int wave = tid >> 5;
    const float* trow = Theta + (size_t)row * NMAT;

    float m = -3.4028235e38f;
    for (int j = tid; j < NMAT; j += 256) m = fmaxf(m, trow[j]);
    #pragma unroll
    for (int off = 16; off > 0; off >>= 1) m = fmaxf(m, __shfl_xor(m, off, 32));
    if ((tid & 31) == 0) red_max[wave] = m;
    __syncthreads();
    m = red_max[0];
    #pragma unroll
    for (int i = 1; i < 8; ++i) m = fmaxf(m, red_max[i]);

    float s = 0.0f;
    for (int j = tid; j < NMAT; j += 256)
        if (j != row) s += __expf(trow[j] - m);
    #pragma unroll
    for (int off = 16; off > 0; off >>= 1) s += __shfl_xor(s, off, 32);
    if ((tid & 31) == 0) red_sum[wave] = s;
    __syncthreads();
    float tot = 0.0f;
    #pragma unroll
    for (int i = 0; i < 8; ++i) tot += red_sum[i];
    tot = (tot > 0.0f) ? tot : 1.0f;
    const float inv = SSTEP / tot;

    float* xrow = X + (size_t)row * NMAT;
    for (int j = tid; j < NMAT; j += 256)
        xrow[j] = (j == row) ? -SSTEP : __expf(trow[j] - m) * inv;
}

// ---------------------------------------------------------------------------
// Kernel 2: P = I + X * (1/6)   (Horner seed)
// ---------------------------------------------------------------------------
__global__ __launch_bounds__(256)
void horner_init_kernel(const float* __restrict__ X, float* __restrict__ P) {
    size_t i = (size_t)blockIdx.x * 256 + threadIdx.x;
    int r = (int)(i / NMAT);
    int c = (int)(i % NMAT);
    P[i] = X[i] * (1.0f / 6.0f) + ((r == c) ? 1.0f : 0.0f);
}

// ---------------------------------------------------------------------------
// bf16x3 split-precision WMMA GEMM:  C = alpha * A * op(B)  [+ I]
//   A,B fp32 in global; hi/lo bf16 split happens during LDS staging.
//   Per K=32 slab per accumulator tile: acc += Ahi*Bhi + Ahi*Blo + Alo*Bhi
//   (3x V_WMMA_F32_16X16X32_BF16, fp32 accumulate  ->  ~fp32-accurate GEMM
//    at 3/8 the issue slots of the 16x16x4 f32 path).
// Tiles: BM=BN=128, BK=32; 8 waves as 4(m) x 2(n); wave owns 32x64 (2x4 v8f).
// ---------------------------------------------------------------------------
template<bool TRANS_B, bool ADD_I>
__global__ __launch_bounds__(256)
void gemm_bf16x3_kernel(const float* __restrict__ A, const float* __restrict__ B,
                        float* __restrict__ C, int Mrows, int Ncols, int K,
                        float alpha) {
    __shared__ unsigned short sa_hi[128 * LDT];
    __shared__ unsigned short sa_lo[128 * LDT];
    __shared__ unsigned short sb_hi[128 * LDT];   // [n][k]
    __shared__ unsigned short sb_lo[128 * LDT];

    const int tid  = threadIdx.x;
    const int lane = tid & 31;
    const int wave = tid >> 5;
    const int wm   = wave >> 1;    // 0..3 : 32-row group
    const int wn   = wave & 1;     // 0..1 : 64-col group
    const int half = lane >> 4;    // bf16 A frag: half0 K in [0,8)u[16,24); half1 [8,16)u[24,32)
    const int lm   = lane & 15;
    const int m0   = blockIdx.y * 128;
    const int n0   = blockIdx.x * 128;

    v8f acc[2][4];
    #pragma unroll
    for (int mt = 0; mt < 2; ++mt)
        #pragma unroll
        for (int nt = 0; nt < 4; ++nt)
            #pragma unroll
            for (int e = 0; e < 8; ++e) acc[mt][nt][e] = 0.0f;

    for (int k0 = 0; k0 < K; k0 += 32) {
        // ---- stage + split A tile 128x32 ----
        {
            const int r  = tid >> 3;          // 0..31
            const int c4 = (tid & 7) * 4;     // 0..28
            #pragma unroll
            for (int i = 0; i < 4; ++i) {
                const int mr = r + i * 32;
                const float* gp = A + (size_t)(m0 + mr) * K + k0 + c4;
                v4f v = *(const v4f*)gp;
                if (k0 + 32 < K) __builtin_prefetch(gp + 32, 0, 3);
                split4_store(sa_hi, sa_lo, mr * LDT + c4, v);
            }
        }
        // ---- stage + split B tile into [n][k] ----
        if (TRANS_B) {
            const int r  = tid >> 3;
            const int c4 = (tid & 7) * 4;
            #pragma unroll
            for (int i = 0; i < 4; ++i) {
                const int nr = r + i * 32;
                const float* gp = B + (size_t)(n0 + nr) * K + k0 + c4;
                v4f v = *(const v4f*)gp;
                if (k0 + 32 < K) __builtin_prefetch(gp + 32, 0, 3);
                split4_store(sb_hi, sb_lo, nr * LDT + c4, v);
            }
        } else {
            const int kr = tid >> 5;          // 0..7
            const int n4 = (tid & 31) * 4;    // 0..124
            #pragma unroll
            for (int i = 0; i < 4; ++i) {
                const int kk = kr + i * 8;
                const float* gp = B + (size_t)(k0 + kk) * Ncols + n0 + n4;
                v4f v = *(const v4f*)gp;
                #pragma unroll
                for (int e = 0; e < 4; ++e) {
                    unsigned short h, l;
                    split1(v[e], h, l);
                    sb_hi[(n4 + e) * LDT + kk] = h;
                    sb_lo[(n4 + e) * LDT + kk] = l;
                }
            }
        }
        __syncthreads();

        // ---- A fragments for this wave's 2 row tiles ----
        v16bf ah[2], al[2];
        #pragma unroll
        for (int mt = 0; mt < 2; ++mt) {
            const int row  = wm * 32 + mt * 16 + lm;
            const int off0 = row * LDT + half * 8;
            ah[mt] = ld_frag(sa_hi, off0, off0 + 16);
            al[mt] = ld_frag(sa_lo, off0, off0 + 16);
        }
        // ---- 4 col tiles x (hi*hi + hi*lo + lo*hi) ----
        #pragma unroll
        for (int nt = 0; nt < 4; ++nt) {
            const int col  = wn * 64 + nt * 16 + lm;
            const int off0 = col * LDT + half * 8;
            v16bf bh = ld_frag(sb_hi, off0, off0 + 16);
            v16bf bl = ld_frag(sb_lo, off0, off0 + 16);
            #pragma unroll
            for (int mt = 0; mt < 2; ++mt) {
                acc[mt][nt] = __builtin_amdgcn_wmma_f32_16x16x32_bf16(
                    false, ah[mt], false, bh, (short)0, acc[mt][nt], false, false);
                acc[mt][nt] = __builtin_amdgcn_wmma_f32_16x16x32_bf16(
                    false, ah[mt], false, bl, (short)0, acc[mt][nt], false, false);
                acc[mt][nt] = __builtin_amdgcn_wmma_f32_16x16x32_bf16(
                    false, al[mt], false, bh, (short)0, acc[mt][nt], false, false);
            }
        }
        __syncthreads();
    }

    // ---- epilogue: alpha (+ identity); C/D layout: VGPR r -> rows r / r+8 ----
    #pragma unroll
    for (int mt = 0; mt < 2; ++mt) {
        #pragma unroll
        for (int nt = 0; nt < 4; ++nt) {
            const int col = n0 + wn * 64 + nt * 16 + lm;
            #pragma unroll
            for (int r = 0; r < 8; ++r) {
                const int row = m0 + wm * 32 + mt * 16 + r + half * 8;
                float v = alpha * acc[mt][nt][r];
                if (ADD_I) v += (row == col) ? 1.0f : 0.0f;
                C[(size_t)row * Ncols + col] = v;
            }
        }
    }
}

// ---------------------------------------------------------------------------
// Launch:  X = S*A_hat - S*I ; M = Taylor-6 Horner ; out = x * M^T
// ---------------------------------------------------------------------------
extern "C" void kernel_launch(void* const* d_in, const int* in_sizes, int n_in,
                              void* d_out, int out_size, void* d_ws, size_t ws_size,
                              hipStream_t stream) {
    const float* x     = (const float*)d_in[0];   // [B, N]
    const float* Theta = (const float*)d_in[1];   // [N, N]
    float* out = (float*)d_out;                   // [B, N]
    const int Brows = in_sizes[0] / NMAT;         // 16384

    const size_t NN = (size_t)NMAT * NMAT;
    float* X  = (float*)d_ws;                     // 16 MB
    float* P0 = X  + NN;                          // 16 MB
    float* P1 = P0 + NN;                          // 16 MB

    dim3 blk(256);

    xprep_kernel<<<NMAT, blk, 0, stream>>>(Theta, X);
    horner_init_kernel<<<(unsigned)(NN / 256), blk, 0, stream>>>(X, P0);

    dim3 gsq(NMAT / 128, NMAT / 128);
    gemm_bf16x3_kernel<false, true><<<gsq, blk, 0, stream>>>(X, P0, P1, NMAT, NMAT, NMAT, 1.0f / 5.0f);
    gemm_bf16x3_kernel<false, true><<<gsq, blk, 0, stream>>>(X, P1, P0, NMAT, NMAT, NMAT, 1.0f / 4.0f);
    gemm_bf16x3_kernel<false, true><<<gsq, blk, 0, stream>>>(X, P0, P1, NMAT, NMAT, NMAT, 1.0f / 3.0f);
    gemm_bf16x3_kernel<false, true><<<gsq, blk, 0, stream>>>(X, P1, P0, NMAT, NMAT, NMAT, 1.0f / 2.0f);
    gemm_bf16x3_kernel<false, true><<<gsq, blk, 0, stream>>>(X, P0, P1, NMAT, NMAT, NMAT, 1.0f);
    // M in P1

    dim3 gout(NMAT / 128, Brows / 128);
    gemm_bf16x3_kernel<true, false><<<gout, blk, 0, stream>>>(x, P1, out, Brows, NMAT, NMAT, 1.0f);
}
